// TransformerModel_34282428956680
// MI455X (gfx1250) — compile-verified, matrix operand order checked by reading, not run on hardware
//
#include <hip/hip_runtime.h>
#include <stdint.h>

// ---------------------------------------------------------------------------
// MI455X (gfx1250, wave32) transformer forward.
// Heavy math: v_wmma_f32_16x16x32_bf16 (fp32 accumulate), 8 WMMAs per wave
// per K-step (32x64 wave tile).  Tile movement: Tensor Data Mover
// (tensor_load_to_lds) with LDS row padding, double-buffered via TENSORcnt.
// Workspace ~117 MB.
// ---------------------------------------------------------------------------

#define DEV static __device__ __forceinline__

typedef __attribute__((ext_vector_type(16))) __bf16 v16bf;
typedef __attribute__((ext_vector_type(8)))  float  v8f;
typedef __attribute__((ext_vector_type(4)))  unsigned int v4u;
typedef __attribute__((ext_vector_type(8)))  int  v8i;
typedef __attribute__((ext_vector_type(4)))  int  v4i;

union FragAB { uint32_t u[8]; v16bf v; };
union FragC  { float    f[8]; v8f   v; };

DEV uint16_t f2bf(float x) {
  union { float f; uint32_t u; } c; c.f = x;
  uint32_t u = c.u + 0x7FFFu + ((c.u >> 16) & 1u);   // round-to-nearest-even
  return (uint16_t)(u >> 16);
}
DEV float bf2f(uint16_t b) {
  union { uint32_t u; float f; } c; c.u = ((uint32_t)b) << 16;
  return c.f;
}
DEV uint32_t pack2(float lo, float hi) {
  return (uint32_t)f2bf(lo) | ((uint32_t)f2bf(hi) << 16);
}

// ------------------------------ model dims ---------------------------------
#define NL 8
#define NH 12
#define DM 768
#define DK 64
#define DH 3072
#define BB 2
#define TT 1024
#define NDICT 32000
#define ROWS (BB*TT)            // 2048

// ------------------------------ tile config --------------------------------
#define BM 128
#define BK 32
#define LDK 34                  // padded K stride in LDS (bf16): 17 dwords/row

enum { MODE_BF16OUT = 0, MODE_SCORES = 1, MODE_ATTNV = 2,
       MODE_MLP0 = 3, MODE_ADD_H = 4, MODE_UNEMBED = 5, MODE_BF16OUT_T = 6 };

struct GArgs {
  const uint16_t* A;     // bf16 A base
  const float*    Bw;    // fp32 weight matrix [K x ldb] row-major
  const uint16_t* Bh;    // bf16 B base (scores: K matrix, k-contiguous rows)
  const uint16_t* V;     // bf16 v tensor, transposed [NH][DM][ROWS] (attnv)
  uint16_t*       Obf;   // bf16 output
  float*          Of;    // fp32 output (h accumulator or logits)
  uint16_t*       Obf2;  // bf16 shadow of h
  const float*    bias;
  const float*    invRs; // 1/(rowsum+eps), [B*H*T]
  int M, N, K;
  int strideAm;          // A row stride (elements)
  int ldb;               // fp32 B row stride
  int ldc;               // output row stride
  long long batchA, batchB, batchC;
};

// ---------------------------------------------------------------------------
// TDM: issue a 2D bf16 tile load global -> LDS.
// D# per ISA ch.8: group0 = {count, lds_addr, global_addr, type=2},
// group1 = {data_size=2B, pad_enable, pad_interval=16 dwords, pad_amount=1
// dword (row pitch 17 dwords == LDK bf16), tensor dims (zero-fill OOB),
// tile dims, 48-bit dim0 stride}.  Tracked by TENSORcnt.
// ---------------------------------------------------------------------------
DEV void tdm_load_bf16(uint32_t lds_off, const uint16_t* gptr,
                       uint32_t tensor_d0, uint32_t tensor_d1,
                       uint32_t tile_d0, uint32_t tile_d1,
                       uint64_t stride0_elems) {
  const uint64_t ga = (uint64_t)gptr;
  v4u g0;
  g0[0] = 1u;                                         // count=1, user mode
  g0[1] = lds_off;                                    // LDS byte address
  g0[2] = (uint32_t)ga;                               // global_addr[31:0]
  g0[3] = (uint32_t)((ga >> 32) & 0x1FFFFFFu) | (2u << 30);  // addr[56:32]|type=2
  v8i g1;
  g1[0] = (int)((1u << 16) | (1u << 20) | (3u << 22));  // 2B elems, pad 16dw+1dw
  g1[1] = (int)((tensor_d0 & 0xFFFFu) << 16);
  g1[2] = (int)((tensor_d0 >> 16) | ((tensor_d1 & 0xFFFFu) << 16));
  g1[3] = (int)((tensor_d1 >> 16) | (tile_d0 << 16));
  g1[4] = (int)(tile_d1 & 0xFFFFu);                   // tile_dim2 = 0
  g1[5] = (int)(stride0_elems & 0xFFFFFFFFu);
  g1[6] = (int)((stride0_elems >> 32) & 0xFFFFu);     // dim1_stride = 0
  g1[7] = 0;
  const v4i gz = {0, 0, 0, 0};
#if __clang_major__ >= 23
  const v8i gz8 = {0, 0, 0, 0, 0, 0, 0, 0};
  __builtin_amdgcn_tensor_load_to_lds(g0, g1, gz, gz, gz8, 0);
#else
  __builtin_amdgcn_tensor_load_to_lds(g0, g1, gz, gz, 0);
#endif
}

// Load one 16x32 bf16 fragment from an LDS tile stored [row][K] (stride LDK).
// Per ISA 7.12.2 (16-bit A-matrix 16x32): lane L holds row L%16; element pair
// p (p=0..3) covers K = 8*(L/16)+2p(+1); p=4..7 covers K = 16+8*(L/16)+2(p-4).
DEV void frag_from_lds(const uint16_t* base, int lane, FragAB& f) {
  const uint16_t* p = base + (lane & 15) * LDK;
  const int h8 = (lane >> 4) * 8;
#pragma unroll
  for (int j = 0; j < 8; ++j) {
    const int k = (j < 4) ? (h8 + 2 * j) : (16 + h8 + 2 * (j - 4));
    f.u[j] = *(const uint32_t*)(p + k);
  }
}

template <int MODE>
__global__ __launch_bounds__(256) void gemm_kernel(GArgs g) {
  constexpr bool TDM_A = (MODE != MODE_ATTNV);
  constexpr bool TDM_B = (MODE == MODE_SCORES || MODE == MODE_ATTNV);
  constexpr int  BNt   = (MODE == MODE_BF16OUT) ? 64 : 128;  // block tile width
  constexpr int  TJn   = BNt / 32;     // B-fragments per wave (2 or 4)

  __shared__ __align__(16) uint16_t As[2][BM * LDK];
  __shared__ __align__(16) uint16_t Bs[2][BNt * LDK];

  const int tid  = threadIdx.x;
  const int z    = blockIdx.z;
  const int m0   = blockIdx.y * BM;
  const int n0   = blockIdx.x * BNt;
  const int lane = tid & 31;
  const int w    = tid >> 5;           // 8 waves
  const int wm   = w >> 1;             // 0..3 -> 32-row strip
  const int wn   = w & 1;              // 0..1 -> (BNt/2)-col strip

  const uint16_t* Abase = g.A;
  const uint16_t* Bhb   = g.Bh;
  const float*    Bw    = g.Bw;
  int bz = 0;
  if constexpr (MODE == MODE_SCORES) {
    const int hh = z % NH; bz = z / NH;
    Abase = g.A  + ((long long)hh * ROWS + (long long)bz * TT) * DK;  // q
    Bhb   = g.Bh + ((long long)hh * ROWS + (long long)bz * TT) * DK;  // k
  } else if constexpr (MODE == MODE_ATTNV) {
    bz = z;
    Abase = g.A + (long long)bz * (long long)NH * TT * TT;            // P[b]
  } else {
    Abase = g.A + (long long)z * g.batchA;
    if (g.Bw) Bw = g.Bw + (long long)z * g.batchB;
  }

  // Issue the TDM tile loads for K-offset k0 into LDS buffer `buf`.
  auto issue_tiles = [&](int k0, int buf) {
    if constexpr (TDM_A) {
      tdm_load_bf16((uint32_t)(uintptr_t)&As[buf][0],
                    Abase + (long long)m0 * g.strideAm + k0,
                    (uint32_t)(g.K - k0), (uint32_t)(g.M - m0),  // OOB rows -> 0
                    BK, BM, (uint64_t)g.strideAm);
    }
    if constexpr (MODE == MODE_SCORES) {
      tdm_load_bf16((uint32_t)(uintptr_t)&Bs[buf][0],
                    Bhb + (long long)n0 * DK + k0,
                    (uint32_t)(DK - k0), (uint32_t)(g.N - n0),
                    BK, BNt, (uint64_t)DK);
    } else if constexpr (MODE == MODE_ATTNV) {
      const int hh = k0 >> 10, o0 = k0 & (TT - 1);
      tdm_load_bf16((uint32_t)(uintptr_t)&Bs[buf][0],
                    g.V + ((long long)(hh * DM + n0)) * ROWS +
                        (long long)bz * TT + o0,
                    (uint32_t)(TT - o0), (uint32_t)(DM - n0),
                    BK, BNt, (uint64_t)ROWS);
    }
  };

  FragC acc[2][TJn];
#pragma unroll
  for (int i = 0; i < 2; ++i)
#pragma unroll
    for (int j = 0; j < TJn; ++j)
#pragma unroll
      for (int e = 0; e < 8; ++e) acc[i][j].f[e] = 0.f;

  if (w == 0) issue_tiles(0, 0);       // prologue DMA into buffer 0

  const int steps = g.K / BK;
  for (int it = 0; it < steps; ++it) {
    const int k0  = it * BK;
    const int cur = it & 1;

    // ---- manual fills (only paths TDM cannot express) ----
    if constexpr (MODE == MODE_ATTNV) {
      // A = P scaled per-row by 1/rowsum (head taken from K position)
      const int hh = k0 >> 10, o0 = k0 & (TT - 1);
      const uint16_t* Pb = Abase + (long long)hh * TT * TT;
      const float* irs = g.invRs + (long long)(bz * NH + hh) * TT;
      for (int i = tid; i < BM * (BK / 4); i += 256) {
        const int r = i >> 3, c = i & 7;             // c: group of 4 bf16
        const int gm = m0 + r;
        const uint64_t u =
            *(const uint64_t*)(Pb + (long long)gm * TT + o0 + 4 * c);
        const float s = irs[gm];
        uint32_t* d = (uint32_t*)&As[cur][r * LDK + 4 * c];
        d[0] = pack2(bf2f((uint16_t)u) * s, bf2f((uint16_t)(u >> 16)) * s);
        d[1] = pack2(bf2f((uint16_t)(u >> 32)) * s, bf2f((uint16_t)(u >> 48)) * s);
      }
    }
    if constexpr (!TDM_B) {
      // fp32 weights -> bf16, transposed into LDS [n][k] (float4 loads)
      for (int i = tid; i < (BNt * BK) / 4; i += 256) {
        const int k  = i / (BNt / 4);
        const int n4 = (i % (BNt / 4)) * 4;
        const float4 f =
            *(const float4*)(Bw + (long long)(k0 + k) * g.ldb + n0 + n4);
        Bs[cur][(n4 + 0) * LDK + k] = f2bf(f.x);
        Bs[cur][(n4 + 1) * LDK + k] = f2bf(f.y);
        Bs[cur][(n4 + 2) * LDK + k] = f2bf(f.z);
        Bs[cur][(n4 + 3) * LDK + k] = f2bf(f.w);
      }
      if (k0 + BK < g.K)   // prefetch next weight tile (global_prefetch_b8)
        __builtin_prefetch(&Bw[(long long)(k0 + BK) * g.ldb + n0 + (tid & (BNt - 1))],
                           0, 0);
    }

    if (w == 0) __builtin_amdgcn_s_wait_tensorcnt(0);  // DMA for `cur` done
    __syncthreads();
    if (w == 0 && it + 1 < steps) issue_tiles(k0 + BK, cur ^ 1);  // overlap

    // ---------------- 8 (or 4) WMMAs per wave ----------------
    FragAB a[2], b[TJn];
#pragma unroll
    for (int ti = 0; ti < 2; ++ti)
      frag_from_lds(&As[cur][(wm * 32 + ti * 16) * LDK], lane, a[ti]);
#pragma unroll
    for (int tj = 0; tj < TJn; ++tj)
      frag_from_lds(&Bs[cur][(wn * (BNt / 2) + tj * 16) * LDK], lane, b[tj]);
#pragma unroll
    for (int ti = 0; ti < 2; ++ti)
#pragma unroll
      for (int tj = 0; tj < TJn; ++tj)
        acc[ti][tj].v = __builtin_amdgcn_wmma_f32_16x16x32_bf16(
            false, a[ti].v, false, b[tj].v, (short)0, acc[ti][tj].v,
            false, false);
  }

  // ---------------- epilogue ----------------
  // C layout (ISA 7.12.2): VGPR j -> M = 8*(lane/16)+j, N = lane%16.
#pragma unroll
  for (int ti = 0; ti < 2; ++ti) {
#pragma unroll
    for (int tj = 0; tj < TJn; ++tj) {
      const int col  = n0 + wn * (BNt / 2) + tj * 16 + (lane & 15);
      const int row0 = m0 + wm * 32 + ti * 16 + ((lane >> 4) * 8);
#pragma unroll
      for (int j = 0; j < 8; ++j) {
        const int row = row0 + j;
        if (row >= g.M) continue;
        float v = acc[ti][tj].f[j];
        if constexpr (MODE == MODE_BF16OUT) {
          g.Obf[(long long)z * g.batchC + (long long)row * g.ldc + col] = f2bf(v);
        } else if constexpr (MODE == MODE_BF16OUT_T) {
          // transposed store: v_bfT[head][col][row]
          g.Obf[(long long)z * g.batchC + (long long)col * ROWS + row] = f2bf(v);
        } else if constexpr (MODE == MODE_SCORES) {
          v = __expf(fminf(v, 50.f));
          if (col > row) v = 0.f;                      // causal mask
          g.Obf[(long long)z * g.batchC + (long long)row * g.ldc + col] = f2bf(v);
        } else if constexpr (MODE == MODE_ATTNV || MODE == MODE_ADD_H) {
          const long long r = (MODE == MODE_ATTNV)
                                  ? (long long)(bz * TT + row) : (long long)row;
          const long long idx = r * DM + col;
          const float nv = g.Of[idx] + v;
          g.Of[idx]   = nv;
          g.Obf2[idx] = f2bf(nv);
        } else if constexpr (MODE == MODE_MLP0) {
          v = fmaxf(v + g.bias[col], 0.f);
          g.Obf[(long long)row * DH + col] = f2bf(v);
        } else if constexpr (MODE == MODE_UNEMBED) {
          g.Of[(long long)row * g.N + col] = v + g.bias[col];
        }
      }
    }
  }
}

// ------------------------ embedding + positional ---------------------------
__global__ __launch_bounds__(256) void embed_kernel(const int* __restrict__ x,
                                                    const float* __restrict__ emb,
                                                    const float* __restrict__ pos,
                                                    float* __restrict__ h,
                                                    uint16_t* __restrict__ h_bf) {
  const long long i = (long long)blockIdx.x * 256 + threadIdx.x;
  if (i >= (long long)ROWS * DM) return;
  const int row = (int)(i / DM), d = (int)(i % DM);
  const int t = row & (TT - 1);
  const float v = emb[(long long)x[row] * DM + d] + pos[(long long)t * DM + d];
  h[i] = v;
  h_bf[i] = f2bf(v);
}

// ---------------- row sums of unnormalized attention ----------------------
__global__ __launch_bounds__(256) void rowsum_kernel(const uint16_t* __restrict__ P,
                                                     float* __restrict__ invRs,
                                                     int rows) {
  const int row  = blockIdx.x * (blockDim.x >> 5) + (threadIdx.x >> 5);
  const int lane = threadIdx.x & 31;
  if (row >= rows) return;
  const uint16_t* p = P + (long long)row * TT;
  float s = 0.f;
  for (int i = lane; i < TT; i += 32) s += bf2f(p[i]);
#pragma unroll
  for (int off = 16; off > 0; off >>= 1) s += __shfl_down(s, off, 32);
  if (lane == 0) invRs[row] = 1.0f / (s + 1e-10f);
}

// ---------------------------------------------------------------------------
extern "C" void kernel_launch(void* const* d_in, const int* in_sizes, int n_in,
                              void* d_out, int out_size, void* d_ws, size_t ws_size,
                              hipStream_t stream) {
  (void)in_sizes; (void)n_in; (void)out_size; (void)ws_size;
  const int*   x     = (const int*)d_in[0];
  // d_in[1] = last_only (reference uses 1; output shape already reflects it)
  const float* emb   = (const float*)d_in[2];
  const float* pos   = (const float*)d_in[3];
  const float* wq    = (const float*)d_in[4];
  const float* wk    = (const float*)d_in[5];
  const float* wv    = (const float*)d_in[6];
  const float* mlp0  = (const float*)d_in[7];
  const float* mlpb  = (const float*)d_in[8];
  const float* mlp1  = (const float*)d_in[9];
  const float* unemb = (const float*)d_in[10];
  const float* bias  = (const float*)d_in[11];
  float* out = (float*)d_out;

  // ---- workspace carve-up (all 256B-aligned chunks), ~117 MB total ----
  char* wsb = (char*)d_ws;
  size_t off = 0;
  auto carve = [&](size_t bytes) -> void* {
    void* p = wsb + off;
    off += (bytes + 255) & ~(size_t)255;
    return p;
  };
  float*    h     = (float*)   carve((size_t)ROWS * DM * 4);
  uint16_t* h_bf  = (uint16_t*)carve((size_t)ROWS * DM * 2);
  uint16_t* q_bf  = (uint16_t*)carve((size_t)NH * ROWS * DK * 2);
  uint16_t* k_bf  = (uint16_t*)carve((size_t)NH * ROWS * DK * 2);
  uint16_t* v_bfT = (uint16_t*)carve((size_t)NH * DM * ROWS * 2);
  uint16_t* Pm    = (uint16_t*)carve((size_t)BB * NH * TT * TT * 2);
  float*    invRs = (float*)   carve((size_t)BB * NH * TT * 4);
  uint16_t* y_bf  = (uint16_t*)carve((size_t)ROWS * DH * 2);

  const dim3 blk(256);

  // h = embedding[x] + pos
  embed_kernel<<<dim3((ROWS * DM + 255) / 256), blk, 0, stream>>>(x, emb, pos, h, h_bf);

  for (int l = 0; l < NL; ++l) {
    // ---- q = h @ wq[l,h], k = h @ wk[l,h]  (batched over heads, BN=64) ----
    GArgs gq{};
    gq.A = h_bf; gq.Bw = wq + (long long)l * NH * DM * DK; gq.Obf = q_bf;
    gq.M = ROWS; gq.N = DK; gq.K = DM;
    gq.strideAm = DM; gq.ldb = DK; gq.ldc = DK;
    gq.batchA = 0; gq.batchB = (long long)DM * DK; gq.batchC = (long long)ROWS * DK;
    gemm_kernel<MODE_BF16OUT><<<dim3(DK / 64, ROWS / BM, NH), blk, 0, stream>>>(gq);

    GArgs gk = gq;
    gk.Bw = wk + (long long)l * NH * DM * DK; gk.Obf = k_bf;
    gemm_kernel<MODE_BF16OUT><<<dim3(DK / 64, ROWS / BM, NH), blk, 0, stream>>>(gk);

    // ---- v = h @ wv[l,h], stored transposed [h][dim][row] for TDM reuse ----
    GArgs gvp{};
    gvp.A = h_bf; gvp.Bw = wv + (long long)l * NH * DM * DM; gvp.Obf = v_bfT;
    gvp.M = ROWS; gvp.N = DM; gvp.K = DM;
    gvp.strideAm = DM; gvp.ldb = DM; gvp.ldc = DM;
    gvp.batchA = 0; gvp.batchB = (long long)DM * DM; gvp.batchC = (long long)DM * ROWS;
    gemm_kernel<MODE_BF16OUT_T><<<dim3(DM / 128, ROWS / BM, NH), blk, 0, stream>>>(gvp);

    // ---- P = exp(min(q k^T, 50)) * causal  (per b,h) ----
    GArgs gs{};
    gs.A = q_bf; gs.Bh = k_bf; gs.Obf = Pm;
    gs.M = TT; gs.N = TT; gs.K = DK;
    gs.strideAm = DK; gs.ldc = TT; gs.batchC = (long long)TT * TT;
    gemm_kernel<MODE_SCORES><<<dim3(TT / 128, TT / BM, BB * NH), blk, 0, stream>>>(gs);

    // ---- row sums -> invRs ----
    rowsum_kernel<<<dim3(BB * NH * TT / 8), blk, 0, stream>>>(Pm, invRs, BB * NH * TT);

    // ---- h += sum_h (P/rowsum) @ v   (one K=12288 GEMM per b) ----
    GArgs ga{};
    ga.A = Pm; ga.V = v_bfT; ga.invRs = invRs; ga.Of = h; ga.Obf2 = h_bf;
    ga.M = TT; ga.N = DM; ga.K = NH * TT;
    gemm_kernel<MODE_ATTNV><<<dim3(DM / 128, TT / BM, BB), blk, 0, stream>>>(ga);

    // ---- y = relu(h @ mlp0[l] + mlpb[l]) ----
    GArgs g0{};
    g0.A = h_bf; g0.Bw = mlp0 + (long long)l * DM * DH;
    g0.bias = mlpb + (long long)l * DH; g0.Obf = y_bf;
    g0.M = ROWS; g0.N = DH; g0.K = DM;
    g0.strideAm = DM; g0.ldb = DH; g0.ldc = DH;
    gemm_kernel<MODE_MLP0><<<dim3(DH / 128, ROWS / BM, 1), blk, 0, stream>>>(g0);

    // ---- h += y @ mlp1[l] ----
    GArgs g1{};
    g1.A = y_bf; g1.Bw = mlp1 + (long long)l * DH * DM;
    g1.Of = h; g1.Obf2 = h_bf;
    g1.M = ROWS; g1.N = DM; g1.K = DH;
    g1.strideAm = DH; g1.ldb = DM; g1.ldc = DM;
    gemm_kernel<MODE_ADD_H><<<dim3(DM / 128, ROWS / BM, 1), blk, 0, stream>>>(g1);
  }

  // ---- logits = h[:, -1, :] @ unembedding + bias  (M = 2) ----
  GArgs gu{};
  gu.A = h_bf + (long long)(TT - 1) * DM;   // rows 1023, 2047 of h_bf
  gu.strideAm = TT * DM;
  gu.Bw = unemb; gu.bias = bias; gu.Of = out;
  gu.M = BB; gu.N = NDICT; gu.K = DM;
  gu.ldb = NDICT; gu.ldc = NDICT;
  gemm_kernel<MODE_UNEMBED><<<dim3(NDICT / 128, 1, 1), blk, 0, stream>>>(gu);
}
